// QuantumComputerLLM_48017734369388
// MI455X (gfx1250) — compile-verified, matrix operand order checked by reading, not run on hardware
//
#include <hip/hip_runtime.h>
#include <stdint.h>

// ---------------------------------------------------------------------------
// Model constants (match reference)
// ---------------------------------------------------------------------------
constexpr int Bc   = 4;
constexpr int Sc   = 512;
constexpr int Dc   = 512;
constexpr int Hc   = 8;
constexpr int Lc   = 6;
constexpr int DFFc = 2048;
constexpr int VOCc = 50257;
constexpr int DKc  = Dc / Hc;     // 64
constexpr int Mc   = Bc * Sc;     // 2048 rows of activations
constexpr int ZHc  = Bc * Hc;     // 32 attention batches

typedef __bf16 v16bf __attribute__((ext_vector_type(16)));
typedef float  v8f   __attribute__((ext_vector_type(8)));
typedef unsigned int v4u __attribute__((ext_vector_type(4)));
typedef int          v8i __attribute__((ext_vector_type(8)));
typedef int          v4i __attribute__((ext_vector_type(4)));

#if defined(__gfx1250__) && __has_builtin(__builtin_amdgcn_tensor_load_to_lds)
#define HAVE_TDM 1
#else
#define HAVE_TDM 0
#endif

static __device__ __forceinline__ uint16_t f2bf(float f) {
    uint32_t u = __float_as_uint(f);
    u += 0x7FFFu + ((u >> 16) & 1u);          // round-to-nearest-even
    return (uint16_t)(u >> 16);
}

#if HAVE_TDM
// TDM 2-D tile load: rows x 32 bf16 tile, global row stride = stride_elems,
// into LDS at byte offset lds_off with an 80B padded row pitch
// (pad 4 DWORDs after every 16 DWORDs => 64B data + 16B pad per row).
static __device__ __forceinline__ void tdm_load_2d(uint32_t lds_off,
                                                   const uint16_t* gaddr,
                                                   uint32_t stride_elems,
                                                   uint32_t rows) {
    uint64_t ga = (uint64_t)(uintptr_t)gaddr;
    v4u g0;
    g0.x = 1u;                                                  // count = 1 (valid D#)
    g0.y = lds_off;                                             // lds_addr (bytes)
    g0.z = (uint32_t)ga;                                        // global_addr[31:0]
    g0.w = (uint32_t)((ga >> 32) & 0x01FFFFFFu) | (2u << 30);   // ga[56:32] | type=2
    v8i g1;
    g1[0] = (int)((1u << 16)        // data_size = 2 bytes
                | (1u << 20)        // pad_enable
                | (3u << 22)        // pad_interval: 16 DWORDs
                | (3u << 25));      // pad_amount: 4 DWORDs
    g1[1] = (int)(32u << 16);       // tensor_dim0 = 32 (lo16 in bits 63:48)
    g1[2] = (int)(rows << 16);      // tensor_dim1 lo16 (bits 95:80)
    g1[3] = (int)(32u << 16);       // tile_dim0 = 32 (bits 127:112)
    g1[4] = (int)rows;              // tile_dim1 (bits 143:128), tile_dim2 = 0
    g1[5] = (int)stride_elems;      // tensor_dim0_stride[31:0]
    g1[6] = 0;
    g1[7] = 0;
    v4i zz = {0, 0, 0, 0};
#if __clang_major__ >= 23
    v8i z8 = {0, 0, 0, 0, 0, 0, 0, 0};
    __builtin_amdgcn_tensor_load_to_lds(g0, g1, zz, zz, z8, 0);
#else
    __builtin_amdgcn_tensor_load_to_lds(g0, g1, zz, zz, 0);
#endif
}
#endif

// ---------------------------------------------------------------------------
// Generic batched bf16 GEMM, f32 accumulate, V_WMMA_F32_16X16X32_BF16.
//   C[z] (MxN) = A[z] (MxK) * op(B[z]) + bias
//   OPB==0: B is [K,N] row-major ; OPB==1: B is [N,K] row-major (A*B^T)
// Assumes M % 128 == 0, K % 32 == 0, lda even; OPB==1 requires ldb % 8 == 0
// and full N tiles (true at all call sites). N may be ragged for OPB==0.
// Block tile 128x128, BK=32, 256 threads (8 wave32s); wave tile 32x64
// (2x4 accumulators -> 8 WMMAs per K-step).  Double-buffered LDS; A tile
// (and B tile for OPB==1) staged by the Tensor Data Mover.
// ---------------------------------------------------------------------------
union FragAB  { struct { uint4 lo, hi; } q; v16bf v; };
union Pack128 { uint4 v; uint16_t h[8]; };

template <int OPB, bool HASC, bool HASCB, bool HASBIAS>
__global__ __launch_bounds__(256)
void gemm_bf16_wmma(const uint16_t* __restrict__ A, int lda, long long strideA,
                    const uint16_t* __restrict__ Bm, int ldb, long long strideB,
                    const float* __restrict__ bias,
                    float* __restrict__ C, uint16_t* __restrict__ Cb,
                    int ldc, long long strideC,
                    int M, int N, int K) {
    constexpr int STR    = 40;            // u16 LDS row pitch (80 B)
    constexpr int A_TILE = 128 * STR;     // u16 elems
    constexpr int B_TILE = 128 * STR;
    constexpr int BUF    = A_TILE + B_TILE;
    __shared__ __align__(16) uint16_t smem[2 * BUF];   // 40 KB of 320 KB WGP LDS

    const int t    = threadIdx.x;
    const int lane = t & 31;
    const int wave = t >> 5;
    const int wm   = wave >> 1;           // 0..3 (M sub-tile)
    const int wn   = wave & 1;            // 0..1 (N sub-tile)
    const int z    = blockIdx.z;
    const int row0 = blockIdx.y * 128;
    const int col0 = blockIdx.x * 128;

    A  += (size_t)z * (size_t)strideA;
    Bm += (size_t)z * (size_t)strideB;
    const size_t zc = (size_t)z * (size_t)strideC;

    // Guard-free B staging when the N-tile is full and rows are 16B-alignable;
    // the ragged / odd-stride (all Wout tiles; P*V with N=64) case takes clamped loads.
    const bool fastB = (OPB == 0) && (col0 + 128 <= N) && ((ldb & 7) == 0);

    // ---- tile staging (writes buffer `buf` for k-offset k0) ----
    auto stage = [&](int buf, int k0) {
        const int ao = buf * BUF;             // A tile base (u16 index)
        const int bo = ao + A_TILE;           // B tile base
        // ---------------- A tile: 128 rows x 32 bf16 ----------------
#if HAVE_TDM
        if (wave == 0) {
            tdm_load_2d((uint32_t)(uintptr_t)&smem[ao],
                        A + (size_t)row0 * lda + k0, (uint32_t)lda, 128u);
        }
#else
#pragma unroll
        for (int i = 0; i < 8; ++i) {
            int u = t + i * 256;
            int r = u >> 4, c2 = u & 15;
            uint32_t val = *(const uint32_t*)(A + (size_t)(row0 + r) * lda + k0 + c2 * 2);
            *(uint32_t*)&smem[ao + r * STR + c2 * 2] = val;
        }
#endif
        // ---------------- B tile -> LDS as [n][k] ----------------
        if constexpr (OPB == 1) {
            // B is [N,K]: rows are already in [n][k] orientation.
#if HAVE_TDM
            if (wave == 0) {
                tdm_load_2d((uint32_t)(uintptr_t)&smem[bo],
                            Bm + (size_t)col0 * ldb + k0, (uint32_t)ldb, 128u);
            }
#else
#pragma unroll
            for (int i = 0; i < 8; ++i) {
                int e = t + i * 256;
                int n = e >> 4, c2 = e & 15;
                uint32_t val = *(const uint32_t*)(Bm + (size_t)(col0 + n) * ldb + k0 + c2 * 2);
                *(uint32_t*)&smem[bo + n * STR + c2 * 2] = val;
            }
#endif
        } else if (fastB) {
            // 2x uint4 per thread (8 bf16 each), transpose-scatter to LDS.
#pragma unroll
            for (int i = 0; i < 2; ++i) {
                int u = t + i * 256;           // 512 uint4 total
                int k = u >> 4, q = u & 15;    // row k (0..31), 8-col group q
                Pack128 p;
                p.v = *(const uint4*)(Bm + (size_t)(k0 + k) * ldb + col0 + q * 8);
#pragma unroll
                for (int j = 0; j < 8; ++j)
                    smem[bo + (q * 8 + j) * STR + k] = p.h[j];
            }
        } else {
            // Ragged / odd-stride path: clamped indices, no divergent loads.
#pragma unroll
            for (int i = 0; i < 16; ++i) {
                int e = t + i * 256;
                int k = e >> 7, n = e & 127;
                bool ok = (col0 + n) < N;
                size_t idx = ok ? ((size_t)(k0 + k) * ldb + col0 + n) : 0;
                uint16_t val = Bm[idx];
                smem[bo + n * STR + k] = ok ? val : (uint16_t)0;
            }
        }
    };

    v8f acc[2][4] = {};
    const int mrow = lane & 15;
    const int koff = (lane & 16) ? 8 : 0;     // ISA 16-bit A/B lane K-offset pattern

    // ---- software pipeline: stage(0) ; [stage(next) | compute(cur)] ... ----
    stage(0, 0);
#if HAVE_TDM
    __builtin_amdgcn_s_wait_tensorcnt(0);
#endif
    __syncthreads();

    const int KT = K >> 5;
    for (int kt = 0; kt < KT; ++kt) {
        const int cur = kt & 1;
        const bool has_next = (kt + 1) < KT;
        if (has_next) stage(cur ^ 1, (kt + 1) << 5);

        const int ao = cur * BUF;
        const int bo = ao + A_TILE;
        FragAB a[2], b[4];
#pragma unroll
        for (int mi = 0; mi < 2; ++mi) {
            int base = ao + (wm * 32 + mi * 16 + mrow) * STR + koff;
            a[mi].q.lo = *(const uint4*)&smem[base];
            a[mi].q.hi = *(const uint4*)&smem[base + 16];
        }
#pragma unroll
        for (int ni = 0; ni < 4; ++ni) {
            int base = bo + (wn * 64 + ni * 16 + mrow) * STR + koff;
            b[ni].q.lo = *(const uint4*)&smem[base];
            b[ni].q.hi = *(const uint4*)&smem[base + 16];
        }
#pragma unroll
        for (int mi = 0; mi < 2; ++mi)
#pragma unroll
            for (int ni = 0; ni < 4; ++ni)
                acc[mi][ni] = __builtin_amdgcn_wmma_f32_16x16x32_bf16(
                    false, a[mi].v, false, b[ni].v,
                    (short)0, acc[mi][ni], false, false);

        if (has_next) {
#if HAVE_TDM
            __builtin_amdgcn_s_wait_tensorcnt(0);
#endif
            __syncthreads();
        }
    }

    // ---- epilogue: C layout per ISA (VGPR e -> M = e | e+8; N = lane&15) ----
    const int mhi = (lane & 16) ? 8 : 0;
#pragma unroll
    for (int ni = 0; ni < 4; ++ni) {
        int n = col0 + wn * 64 + ni * 16 + (lane & 15);
        if (n >= N) continue;
        float bv = 0.0f;
        if constexpr (HASBIAS) bv = bias[n];
#pragma unroll
        for (int mi = 0; mi < 2; ++mi) {
            int mbase = row0 + wm * 32 + mi * 16 + mhi;
#pragma unroll
            for (int e = 0; e < 8; ++e) {
                float v = acc[mi][ni][e] + bv;
                size_t off = zc + (size_t)(mbase + e) * ldc + n;
                if constexpr (HASC)  C[off]  = v;
                if constexpr (HASCB) Cb[off] = f2bf(v);
            }
        }
    }
}

// ---------------------------------------------------------------------------
// Elementwise / reduction kernels
// ---------------------------------------------------------------------------
__global__ void embed_kernel(const int* __restrict__ ids,
                             const float* __restrict__ tok,
                             const float* __restrict__ pos,
                             float* __restrict__ x32, uint16_t* __restrict__ x16) {
    long long i = (long long)blockIdx.x * blockDim.x + threadIdx.x;
    if (i >= (long long)Mc * Dc) return;
    int d = (int)(i % Dc);
    long long row = i / Dc;                 // b*S + s
    int s = (int)(row % Sc);
    int id = ids[row];
    float v = tok[(size_t)id * Dc + d] + pos[(size_t)s * Dc + d];
    x32[i] = v;
    x16[i] = f2bf(v);
}

__global__ void cvt_kernel(const float* __restrict__ in, uint16_t* __restrict__ out,
                           long long n) {
    long long i = (long long)blockIdx.x * blockDim.x + threadIdx.x;
    if (i < n) out[i] = f2bf(in[i]);
}

// [B,S,H,DK] -> [B,H,S,DK]
__global__ void perm_to_heads(const uint16_t* __restrict__ in, uint16_t* __restrict__ out) {
    long long i = (long long)blockIdx.x * blockDim.x + threadIdx.x;
    if (i >= (long long)Mc * Dc) return;
    int b = (int)(i / ((long long)Sc * Dc));
    int r = (int)(i % ((long long)Sc * Dc));
    int s = r / Dc, d = r % Dc;
    int h = d / DKc, dk = d % DKc;
    out[(((size_t)b * Hc + h) * Sc + s) * DKc + dk] = in[i];
}

// [B,H,S,DK] -> [B,S,H,DK]
__global__ void perm_from_heads(const uint16_t* __restrict__ in, uint16_t* __restrict__ out) {
    long long i = (long long)blockIdx.x * blockDim.x + threadIdx.x;
    if (i >= (long long)Mc * Dc) return;
    int b = (int)(i / ((long long)Sc * Dc));
    int r = (int)(i % ((long long)Sc * Dc));
    int s = r / Dc, d = r % Dc;
    int h = d / DKc, dk = d % DKc;
    out[i] = in[(((size_t)b * Hc + h) * Sc + s) * DKc + dk];
}

// one block per row of 512 scores; probs out in bf16
__global__ __launch_bounds__(256)
void softmax_kernel(const float* __restrict__ scores, uint16_t* __restrict__ probs,
                    float scale) {
    __shared__ float red[256];
    size_t row = blockIdx.x;
    const float* s = scores + row * Sc;
    uint16_t*    p = probs  + row * Sc;
    int t = threadIdx.x;
    float v0 = s[t] * scale, v1 = s[t + 256] * scale;
    red[t] = fmaxf(v0, v1);
    __syncthreads();
    for (int k = 128; k > 0; k >>= 1) {
        if (t < k) red[t] = fmaxf(red[t], red[t + k]);
        __syncthreads();
    }
    float m = red[0];
    __syncthreads();
    float e0 = __expf(v0 - m), e1 = __expf(v1 - m);
    red[t] = e0 + e1;
    __syncthreads();
    for (int k = 128; k > 0; k >>= 1) {
        if (t < k) red[t] += red[t + k];
        __syncthreads();
    }
    float inv = 1.0f / red[0];
    p[t]       = f2bf(e0 * inv);
    p[t + 256] = f2bf(e1 * inv);
}

// x = LN(X + R) * g + beta ; writes fp32 + bf16.  R may be null (plain LN).
// Safe when x32 aliases R: all reads complete before the first barrier.
__global__ __launch_bounds__(256)
void ln_kernel(const float* __restrict__ X, const float* __restrict__ R,
               const float* __restrict__ g, const float* __restrict__ bta,
               float* __restrict__ x32, uint16_t* __restrict__ x16) {
    __shared__ float rs[256];
    __shared__ float rq[256];
    size_t row = blockIdx.x;
    int t = threadIdx.x;
    const float* xr = X + row * Dc;
    float v0 = xr[t]       + (R ? R[row * Dc + t]       : 0.0f);
    float v1 = xr[t + 256] + (R ? R[row * Dc + t + 256] : 0.0f);
    rs[t] = v0 + v1;
    rq[t] = v0 * v0 + v1 * v1;
    __syncthreads();
    for (int k = 128; k > 0; k >>= 1) {
        if (t < k) { rs[t] += rs[t + k]; rq[t] += rq[t + k]; }
        __syncthreads();
    }
    float mean = rs[0] * (1.0f / Dc);
    float var  = rq[0] * (1.0f / Dc) - mean * mean;
    float rstd = rsqrtf(var + 1e-5f);
    float y0 = (v0 - mean) * rstd * g[t]       + bta[t];
    float y1 = (v1 - mean) * rstd * g[t + 256] + bta[t + 256];
    x32[row * Dc + t]       = y0;  x16[row * Dc + t]       = f2bf(y0);
    x32[row * Dc + t + 256] = y1;  x16[row * Dc + t + 256] = f2bf(y1);
}

__global__ void gelu_kernel(const float* __restrict__ in, uint16_t* __restrict__ out,
                            long long n) {
    long long i = (long long)blockIdx.x * blockDim.x + threadIdx.x;
    if (i >= n) return;
    float x = in[i];
    out[i] = f2bf(0.5f * x * (1.0f + erff(x * 0.70710678118654752f)));
}

// ---------------------------------------------------------------------------
// Host orchestration
// ---------------------------------------------------------------------------
extern "C" void kernel_launch(void* const* d_in, const int* in_sizes, int n_in,
                              void* d_out, int out_size, void* d_ws, size_t ws_size,
                              hipStream_t stream) {
    (void)in_sizes; (void)n_in; (void)out_size; (void)ws_size;

    const int*   ids  = (const int*)d_in[0];
    const float* tok  = (const float*)d_in[1];
    const float* pos  = (const float*)d_in[2];
    const float* Wq   = (const float*)d_in[3];
    const float* bq   = (const float*)d_in[4];
    const float* Wk   = (const float*)d_in[5];
    const float* bk   = (const float*)d_in[6];
    const float* Wv   = (const float*)d_in[7];
    const float* bv   = (const float*)d_in[8];
    const float* Wo   = (const float*)d_in[9];
    const float* bo   = (const float*)d_in[10];
    const float* ln_g = (const float*)d_in[11];
    const float* ln_b = (const float*)d_in[12];
    const float* W1   = (const float*)d_in[13];
    const float* b1   = (const float*)d_in[14];
    const float* W2   = (const float*)d_in[15];
    const float* b2   = (const float*)d_in[16];
    const float* on_g = (const float*)d_in[17];
    const float* on_b = (const float*)d_in[18];
    const float* Wout = (const float*)d_in[19];
    const float* bout = (const float*)d_in[20];
    float* logits = (float*)d_out;

    // ---- workspace layout (256B-aligned slabs) ----
    char* ws = (char*)d_ws;
    size_t off = 0;
    auto alloc = [&](size_t bytes) -> void* {
        void* p = ws + off;
        off = (off + bytes + 255) & ~(size_t)255;
        return p;
    };
    float*    x32   = (float*)   alloc((size_t)Mc * Dc * 4);
    uint16_t* x16   = (uint16_t*)alloc((size_t)Mc * Dc * 2);
    uint16_t* tmp16 = (uint16_t*)alloc((size_t)Mc * Dc * 2);
    uint16_t* qh    = (uint16_t*)alloc((size_t)Mc * Dc * 2);
    uint16_t* kh    = (uint16_t*)alloc((size_t)Mc * Dc * 2);
    uint16_t* vh    = (uint16_t*)alloc((size_t)Mc * Dc * 2);
    float*    sc32  = (float*)   alloc((size_t)ZHc * Sc * Sc * 4);
    uint16_t* pr16  = (uint16_t*)alloc((size_t)ZHc * Sc * Sc * 2);
    uint16_t* oh16  = (uint16_t*)alloc((size_t)Mc * Dc * 2);
    uint16_t* o16   = (uint16_t*)alloc((size_t)Mc * Dc * 2);
    float*    t32   = (float*)   alloc((size_t)Mc * Dc * 4);
    float*    h32   = (float*)   alloc((size_t)Mc * DFFc * 4);
    uint16_t* h16   = (uint16_t*)alloc((size_t)Mc * DFFc * 2);
    uint16_t* wq16  = (uint16_t*)alloc((size_t)Lc * Dc * Dc * 2);
    uint16_t* wk16  = (uint16_t*)alloc((size_t)Lc * Dc * Dc * 2);
    uint16_t* wv16  = (uint16_t*)alloc((size_t)Lc * Dc * Dc * 2);
    uint16_t* wo16  = (uint16_t*)alloc((size_t)Lc * Dc * Dc * 2);
    uint16_t* w116  = (uint16_t*)alloc((size_t)Dc * DFFc * 2);
    uint16_t* w216  = (uint16_t*)alloc((size_t)DFFc * Dc * 2);
    uint16_t* wout16= (uint16_t*)alloc((size_t)Dc * VOCc * 2);

    auto cvt = [&](const float* in, uint16_t* outp, long long n) {
        cvt_kernel<<<(unsigned)((n + 255) / 256), 256, 0, stream>>>(in, outp, n);
    };
    // dispatch to the 4 template instantiations actually used
    auto gemm = [&](const uint16_t* A, int lda, long long sA,
                    const uint16_t* Bp, int ldb, long long sB, int opB,
                    const float* bias, float* C, uint16_t* Cb, int ldc, long long sC,
                    int M, int N, int K, int Z) {
        dim3 grid((unsigned)((N + 127) / 128), (unsigned)(M / 128), (unsigned)Z);
        if (opB == 1)
            gemm_bf16_wmma<1, true, false, false><<<grid, 256, 0, stream>>>(
                A, lda, sA, Bp, ldb, sB, nullptr, C, nullptr, ldc, sC, M, N, K);
        else if (C)
            gemm_bf16_wmma<0, true, false, true><<<grid, 256, 0, stream>>>(
                A, lda, sA, Bp, ldb, sB, bias, C, nullptr, ldc, sC, M, N, K);
        else if (bias)
            gemm_bf16_wmma<0, false, true, true><<<grid, 256, 0, stream>>>(
                A, lda, sA, Bp, ldb, sB, bias, nullptr, Cb, ldc, sC, M, N, K);
        else
            gemm_bf16_wmma<0, false, true, false><<<grid, 256, 0, stream>>>(
                A, lda, sA, Bp, ldb, sB, nullptr, nullptr, Cb, ldc, sC, M, N, K);
    };

    // ---- bf16 weight conversion (every call: deterministic, capture-safe) ----
    cvt(Wq,   wq16,  (long long)Lc * Dc * Dc);
    cvt(Wk,   wk16,  (long long)Lc * Dc * Dc);
    cvt(Wv,   wv16,  (long long)Lc * Dc * Dc);
    cvt(Wo,   wo16,  (long long)Lc * Dc * Dc);
    cvt(W1,   w116,  (long long)Dc * DFFc);
    cvt(W2,   w216,  (long long)DFFc * Dc);
    cvt(Wout, wout16,(long long)Dc * VOCc);

    // ---- embeddings ----
    const long long NBD = (long long)Mc * Dc;
    embed_kernel<<<(unsigned)((NBD + 255) / 256), 256, 0, stream>>>(ids, tok, pos, x32, x16);

    const unsigned permG = (unsigned)((NBD + 255) / 256);
    const long long sHD  = (long long)Sc * DKc;   // per-(b,h) stride in head-major tensors
    const long long sSS  = (long long)Sc * Sc;

    // ---- transformer stack ----
    for (int l = 0; l < Lc; ++l) {
        const size_t wOff = (size_t)l * Dc * Dc;
        // Q, K, V projections (bf16 out), then to head-major layout
        gemm(x16, Dc, 0, wq16 + wOff, Dc, 0, 0, bq + l * Dc,
             nullptr, tmp16, Dc, 0, Mc, Dc, Dc, 1);
        perm_to_heads<<<permG, 256, 0, stream>>>(tmp16, qh);
        gemm(x16, Dc, 0, wk16 + wOff, Dc, 0, 0, bk + l * Dc,
             nullptr, tmp16, Dc, 0, Mc, Dc, Dc, 1);
        perm_to_heads<<<permG, 256, 0, stream>>>(tmp16, kh);
        gemm(x16, Dc, 0, wv16 + wOff, Dc, 0, 0, bv + l * Dc,
             nullptr, tmp16, Dc, 0, Mc, Dc, Dc, 1);
        perm_to_heads<<<permG, 256, 0, stream>>>(tmp16, vh);

        // scores[z] = Q[z] * K[z]^T   (z = b*H + h)
        gemm(qh, DKc, sHD, kh, DKc, sHD, 1, nullptr,
             sc32, nullptr, Sc, sSS, Sc, Sc, DKc, ZHc);
        // softmax(scale * scores) -> bf16 probs
        softmax_kernel<<<(unsigned)(ZHc * Sc), 256, 0, stream>>>(sc32, pr16, 0.125f);
        // O[z] = P[z] * V[z]
        gemm(pr16, Sc, sSS, vh, DKc, sHD, 0, nullptr,
             nullptr, oh16, DKc, sHD, Sc, DKc, Sc, ZHc);
        perm_from_heads<<<permG, 256, 0, stream>>>(oh16, o16);

        // output projection + residual LayerNorm
        gemm(o16, Dc, 0, wo16 + wOff, Dc, 0, 0, bo + l * Dc,
             t32, nullptr, Dc, 0, Mc, Dc, Dc, 1);
        ln_kernel<<<(unsigned)Mc, 256, 0, stream>>>(t32, x32, ln_g + l * Dc, ln_b + l * Dc,
                                                    x32, x16);
    }

    // ---- feed-forward + final LN ----
    gemm(x16, Dc, 0, w116, DFFc, 0, 0, b1, h32, nullptr, DFFc, 0, Mc, DFFc, Dc, 1);
    const long long NH = (long long)Mc * DFFc;
    gelu_kernel<<<(unsigned)((NH + 255) / 256), 256, 0, stream>>>(h32, h16, NH);
    gemm(h16, DFFc, 0, w216, Dc, 0, 0, b2, t32, nullptr, Dc, 0, Mc, Dc, DFFc, 1);
    ln_kernel<<<(unsigned)Mc, 256, 0, stream>>>(t32, nullptr, on_g, on_b, x32, x16);

    // ---- vocab projection: logits (fp32, ragged N=50257) ----
    gemm(x16, Dc, 0, wout16, VOCc, 0, 0, bout,
         logits, nullptr, VOCc, 0, Mc, VOCc, Dc, 1);
}
